// SpectralAttention_19250043421101
// MI455X (gfx1250) — matrix operand order repaired and checked by reading, not compiled, not run on hardware
//
#include <hip/hip_runtime.h>

typedef __attribute__((ext_vector_type(16))) _Float16 v16h;
typedef __attribute__((ext_vector_type(8)))  _Float16 v8h;
typedef __attribute__((ext_vector_type(8)))  float    v8f;
typedef __attribute__((ext_vector_type(4)))  float    v4f;
typedef __attribute__((ext_vector_type(4)))  unsigned v4u;
typedef __attribute__((ext_vector_type(8)))  unsigned v8u;

#define B_   4
#define T_   4096
#define D_   1024
#define H_   16
#define DH_  64
#define BH_  (B_ * H_)         // 64 (b,h) slices
#define SL_  (T_ * DH_)        // 262144 elements per slice
#define NE_  ((long)BH_ * SL_) // 16777216
#define PI2  6.283185307179586f

__device__ __forceinline__ v8f wmma16(v16h a, v16h b, v8f c) {
  return __builtin_amdgcn_wmma_f32_16x16x32_f16(false, a, false, b, (short)0, c,
                                                false, false);
}

// ---------------- CDNA5 async / TDM helpers ----------------------------------
__device__ __forceinline__ unsigned ldsoff(const void* p) {
  // low 32 bits of a generic LDS pointer == byte offset in workgroup LDS
  return (unsigned)(unsigned long long)p;
}
__device__ __forceinline__ void async_b128(unsigned dst, const void* src) {
  asm volatile("global_load_async_to_lds_b128 %0, %1, off"
               :: "v"(dst), "v"(src) : "memory");
}
__device__ __forceinline__ void wait_async0() {
  asm volatile("s_wait_asynccnt 0" ::: "memory");
}

// TDM: 1-row 2D tile, data_size = 2 bytes (f16), contiguous `elems` elements.
// D# packing per CDNA5 ISA §8.3 (group0) / §8.4 (group1).
__device__ __forceinline__ void tdm_load_1row(unsigned lds_addr, const void* src,
                                              unsigned elems) {
  unsigned long long ga = (unsigned long long)src;
  v4u g0;
  g0[0] = 1u;                                   // count=1, user descriptor
  g0[1] = lds_addr;                             // lds_addr        [63:32]
  g0[2] = (unsigned)ga;                         // global_addr lo  [95:64]
  g0[3] = (unsigned)(ga >> 32) | 0x80000000u;   // addr hi | type=2 [127:126]
  v8u g1;
  g1[0] = 0x00010000u;                          // wg_mask=0, data_size=1 (2B)
  g1[1] = (elems & 0xffffu) << 16;              // tensor_dim0[15:0]  @ [63:48]
  g1[2] = (elems >> 16) | (1u << 16);           // tensor_dim0[31:16] | tensor_dim1=1
  g1[3] = (elems & 0xffffu) << 16;              // tile_dim0 @ [127:112]
  g1[4] = 1u;                                   // tile_dim1=1, tile_dim2=0
  g1[5] = elems;                                // tensor_dim0_stride[31:0]
  g1[6] = 0u;
  g1[7] = 0u;
  asm volatile("tensor_load_to_lds %0, %1" :: "s"(g0), "s"(g1) : "memory");
}

// ---------------- fragment loaders (wave32 WMMA f32_16x16x32_f16) -------------
// A (16x32 MxK): lane&15 = row M; K elems {ko..ko+7, ko+16..ko+23}, ko=(lane>>4)*8
__device__ __forceinline__ v16h fragA_f16(const _Float16* p, int ld, int row0,
                                          int k0, int lane) {
  int m  = row0 + (lane & 15);
  int ko = k0 + ((lane >> 4) << 3);
  const _Float16* q = p + (long)m * ld + ko;
  v8h a = *(const v8h*)q;
  v8h b = *(const v8h*)(q + 16);
  v16h f;
#pragma unroll
  for (int i = 0; i < 8; ++i) { f[i] = a[i]; f[i + 8] = b[i]; }
  return f;
}

__device__ __forceinline__ v16h fragA_f32(const float* p, int ld, int row0,
                                          int k0, int lane) {
  int m  = row0 + (lane & 15);
  int ko = k0 + ((lane >> 4) << 3);
  const float* q = p + (long)m * ld + ko;
  v4f a0 = *(const v4f*)q,        a1 = *(const v4f*)(q + 4);
  v4f b0 = *(const v4f*)(q + 16), b1 = *(const v4f*)(q + 20);
  v16h f;
#pragma unroll
  for (int i = 0; i < 4; ++i) {
    f[i]      = (_Float16)a0[i]; f[i + 4]  = (_Float16)a1[i];
    f[i + 8]  = (_Float16)b0[i]; f[i + 12] = (_Float16)b1[i];
  }
  return f;
}

// B (32x16 KxN): lane&15 = col N; K elems {ko..ko+7, ko+16..ko+23}, ko=(lane>>4)*8
// variant where B[k][n] = M[n][k], M row-major f32 (weights stored (N,K))
__device__ __forceinline__ v16h fragB_f32rows(const float* p, int ld, int n0,
                                              int k0, int lane) {
  int n  = n0 + (lane & 15);
  int ks = k0 + ((lane >> 4) << 3);
  const float* q = p + (long)n * ld + ks;
  v16h f;
#pragma unroll
  for (int i = 0; i < 16; i += 4) {
    v4f t = *(const v4f*)(q + i + (i & 8));
    f[i] = (_Float16)t[0]; f[i + 1] = (_Float16)t[1];
    f[i + 2] = (_Float16)t[2]; f[i + 3] = (_Float16)t[3];
  }
  return f;
}

// variant where B[k][n] stored row-major [k][n], ld = leading dim (f16, LDS)
__device__ __forceinline__ v16h fragB_f16strided(const _Float16* p, int ld,
                                                 int k0, int n0, int lane) {
  int n  = n0 + (lane & 15);
  int ks = k0 + ((lane >> 4) << 3);
  const _Float16* q = p + (long)ks * ld + n;
  v16h f;
#pragma unroll
  for (int i = 0; i < 16; ++i) f[i] = q[(long)(i + (i & 8)) * ld];
  return f;
}

// A operand gathered from y stored as [b*H+h][t][dh] (f16), logical A[m][k]
__device__ __forceinline__ v16h fragA_ylayout(const _Float16* y, int m0, int k0,
                                              int lane) {
  int m  = m0 + (lane & 15);
  int b  = m >> 12, t = m & (T_ - 1);
  int ko = k0 + ((lane >> 4) << 3);
  v16h f;
#pragma unroll
  for (int c = 0; c < 2; ++c) {
    int k = ko + c * 16;
    int h = k >> 6, dh = k & 63;
    const _Float16* q = y + (long)(b * H_ + h) * SL_ + (long)t * DH_ + dh;
    v8h a = *(const v8h*)q;
#pragma unroll
    for (int i = 0; i < 8; ++i) f[c * 8 + i] = a[i];
  }
  return f;
}

// ------------------------------ tables ---------------------------------------
__global__ void sa_tables(_Float16* Dc, _Float16* Ds, _Float16* Dsn,
                          float* twc, float* tws) {
  for (int i = threadIdx.x + blockIdx.x * blockDim.x; i < 4096;
       i += blockDim.x * gridDim.x) {
    int a = i >> 6, b = i & 63;
    int p  = (a * b) & 63;
    float ang = (PI2 / 64.0f) * (float)p;
    float c = cosf(ang), s = sinf(ang);
    Dc[i]  = (_Float16)c;
    Ds[i]  = (_Float16)s;
    Dsn[i] = (_Float16)(-s);
    int p2 = (a * b) & 4095;
    float ang2 = (PI2 / 4096.0f) * (float)p2;
    twc[i] = cosf(ang2);
    tws[i] = sinf(ang2);
  }
}

// Hermitian-extended filter with 1/T folded in
__global__ void sa_filter(const float* lg, const float* ph, float* fr, float* fi) {
  int idx = threadIdx.x + blockIdx.x * blockDim.x;
  if (idx >= H_ * 4096) return;
  int h = idx >> 12, k = idx & 4095;
  int src = (k <= 2048) ? k : 4096 - k;
  float sg = (k <= 2048) ? 1.0f : -1.0f;
  float mag = expf(lg[h * 2049 + src]) * (1.0f / 4096.0f);
  float p   = ph[h * 2049 + src];
  fr[idx] = mag * cosf(p);
  fi[idx] = sg * mag * sinf(p);
}

// ------------------- GEMM1: v[b,h,t,dh] = x . Wv^T + bv ----------------------
// Block = 8 m-tiles sharing one 64-wide n-block; W staged via async double buffer.
__global__ __launch_bounds__(256) void sa_gemm1(const float* __restrict__ x,
                                                const float* __restrict__ W,
                                                const float* __restrict__ bias,
                                                _Float16* __restrict__ v_t) {
  __shared__ float sW[2][64 * 32];
  int tid = threadIdx.x, lane = tid & 31, w = tid >> 5;
  int wn = blockIdx.x & 15;
  int m0 = ((blockIdx.x >> 4) * 8 + w) * 16;
  int n0 = wn * 64;
  const float* Wv = W + (long)2 * D_ * D_ + (long)n0 * D_;
  for (int c = tid; c < 512; c += 256) {
    int r = c >> 3, cc = c & 7;
    async_b128(ldsoff(sW[0]) + (r * 32 + cc * 4) * 4, Wv + (long)r * D_ + cc * 4);
  }
  wait_async0();
  __syncthreads();
  v8f acc[4] = {};
  int p = 0;
  for (int kb = 0; kb < D_; kb += 32, p ^= 1) {
    if (kb + 32 < D_) {
      for (int c = tid; c < 512; c += 256) {
        int r = c >> 3, cc = c & 7;
        async_b128(ldsoff(sW[p ^ 1]) + (r * 32 + cc * 4) * 4,
                   Wv + (long)r * D_ + kb + 32 + cc * 4);
      }
    }
    v16h a = fragA_f32(x, D_, m0, kb, lane);
    __builtin_prefetch(x + (long)(m0 + (lane & 15)) * D_ + kb + 64, 0, 0);
#pragma unroll
    for (int s = 0; s < 4; ++s) {
      v16h b = fragB_f32rows(sW[p], 32, s * 16, 0, lane);
      acc[s] = wmma16(a, b, acc[s]);
    }
    wait_async0();
    __syncthreads();
  }
  int mbase = m0 + ((lane >> 4) << 3);
#pragma unroll
  for (int s = 0; s < 4; ++s) {
    int n = n0 + s * 16 + (lane & 15);
    float bv = bias[2 * D_ + n];
    int h = n >> 6, dh = n & 63;
#pragma unroll
    for (int r = 0; r < 8; ++r) {
      int m = mbase + r;
      int b = m >> 12, t = m & (T_ - 1);
      v_t[(long)(b * H_ + h) * SL_ + (long)t * DH_ + dh] =
          (_Float16)(acc[s][r] + bv);
    }
  }
}

// --------- stage1: A[k2][t1*64+dh] = tw(t1,k2) * (D64 . V)  (fwd inner) ------
__global__ __launch_bounds__(256) void sa_stage1(const _Float16* __restrict__ v_t,
                                                 const _Float16* __restrict__ Dc,
                                                 const _Float16* __restrict__ Dsn,
                                                 const float* __restrict__ twc,
                                                 const float* __restrict__ tws,
                                                 _Float16* __restrict__ Ar,
                                                 _Float16* __restrict__ Ai) {
  __shared__ _Float16 sV[2][4096];
  __shared__ _Float16 sDc[4096], sDn[4096];
  int tid = threadIdx.x;
  int bh = blockIdx.x >> 5;
  int jp = blockIdx.x & 31;   // pair of 64-col blocks
  const _Float16* V = v_t + (long)bh * SL_;
  for (int c = tid; c < 512; c += 256) async_b128(ldsoff(sDc) + c * 16, Dc + c * 8);
  for (int c = tid; c < 512; c += 256) async_b128(ldsoff(sDn) + c * 16, Dsn + c * 8);
  for (int c = tid; c < 1024; c += 256) {
    int jb = c >> 9, r = (c >> 3) & 63, cc = c & 7;
    async_b128(ldsoff(sV) + (jb * 4096 + r * 64 + cc * 8) * 2,
               V + (long)r * 4096 + jp * 128 + jb * 64 + cc * 8);
  }
  wait_async0();
  __syncthreads();

  int lane = tid & 31, w = tid >> 5;
  int jb = w >> 2;
  int m0 = (w & 3) * 16;          // k2 rows
  int t1 = jp * 2 + jb;
  int j0 = t1 * 64;
  v8f cr[4] = {}, ci[4] = {};
  for (int kb = 0; kb < 64; kb += 32) {
    v16h ac = fragA_f16(sDc, 64, m0, kb, lane);
    v16h an = fragA_f16(sDn, 64, m0, kb, lane);
#pragma unroll
    for (int s = 0; s < 4; ++s) {
      v16h b = fragB_f16strided(sV[jb], 64, kb, s * 16, lane);
      cr[s] = wmma16(ac, b, cr[s]);   // cos . v
      ci[s] = wmma16(an, b, ci[s]);   // -sin . v
    }
  }
  int mbase = m0 + ((lane >> 4) << 3);
  _Float16* pr = Ar + (long)bh * SL_;
  _Float16* pi = Ai + (long)bh * SL_;
#pragma unroll
  for (int s = 0; s < 4; ++s) {
    int n = j0 + s * 16 + (lane & 15);
#pragma unroll
    for (int r = 0; r < 8; ++r) {
      int k2 = mbase + r;
      float tc = twc[t1 * 64 + k2], ts = tws[t1 * 64 + k2];
      float re = cr[s][r], im = ci[s][r];
      pr[(long)k2 * 4096 + n] = (_Float16)(re * tc + im * ts);
      pi[(long)k2 * 4096 + n] = (_Float16)(im * tc - re * ts);
    }
  }
}

// ---- stage2: X[k2][k1][dh] = filt[h][k2+64k1] * (D64 . A)  (fwd outer) ------
__global__ __launch_bounds__(256) void sa_stage2(const _Float16* __restrict__ Ar,
                                                 const _Float16* __restrict__ Ai,
                                                 const _Float16* __restrict__ Dc,
                                                 const _Float16* __restrict__ Ds,
                                                 const _Float16* __restrict__ Dsn,
                                                 const float* __restrict__ fr,
                                                 const float* __restrict__ fi,
                                                 _Float16* __restrict__ Xr,
                                                 _Float16* __restrict__ Xi) {
  __shared__ _Float16 sAr[2][4096], sAi[2][4096];
  __shared__ _Float16 sDc[4096], sDs[4096], sDn[4096];
  int tid = threadIdx.x;
  long batch0 = (long)blockIdx.x * 2;
  if (tid < 32) {               // wave0 drives the Tensor Data Mover
    tdm_load_1row(ldsoff(sAr), Ar + batch0 * 4096, 8192);
    tdm_load_1row(ldsoff(sAi), Ai + batch0 * 4096, 8192);
    __builtin_amdgcn_s_wait_tensorcnt(0);
  }
  for (int c = tid; c < 512; c += 256) async_b128(ldsoff(sDc) + c * 16, Dc + c * 8);
  for (int c = tid; c < 512; c += 256) async_b128(ldsoff(sDs) + c * 16, Ds + c * 8);
  for (int c = tid; c < 512; c += 256) async_b128(ldsoff(sDn) + c * 16, Dsn + c * 8);
  wait_async0();
  __syncthreads();

  int lane = tid & 31, w = tid >> 5;
  int bl = w >> 2;                // local batch
  int m0 = (w & 3) * 16;          // k1 rows
  long batch = batch0 + bl;
  int bh = (int)(batch >> 6), k2 = (int)(batch & 63);
  int h = bh & (H_ - 1);
  v8f xr[4] = {}, xi[4] = {};
  for (int kb = 0; kb < 64; kb += 32) {          // contract over t1
    v16h aC = fragA_f16(sDc, 64, m0, kb, lane);
    v16h aS = fragA_f16(sDs, 64, m0, kb, lane);
    v16h aN = fragA_f16(sDn, 64, m0, kb, lane);
#pragma unroll
    for (int s = 0; s < 4; ++s) {
      v16h br = fragB_f16strided(sAr[bl], 64, kb, s * 16, lane);
      v16h bi = fragB_f16strided(sAi[bl], 64, kb, s * 16, lane);
      xr[s] = wmma16(aC, br, xr[s]);
      xr[s] = wmma16(aS, bi, xr[s]);
      xi[s] = wmma16(aC, bi, xi[s]);
      xi[s] = wmma16(aN, br, xi[s]);
    }
  }
  int mbase = m0 + ((lane >> 4) << 3);
#pragma unroll
  for (int s = 0; s < 4; ++s) {
    int n = s * 16 + (lane & 15);
#pragma unroll
    for (int r = 0; r < 8; ++r) {
      int k1 = mbase + r;
      int k  = k2 + 64 * k1;
      float cr_ = fr[h * 4096 + k], ci_ = fi[h * 4096 + k];
      float re = xr[s][r], im = xi[s][r];
      Xr[batch * 4096 + (long)k1 * 64 + n] = (_Float16)(re * cr_ - im * ci_);
      Xi[batch * 4096 + (long)k1 * 64 + n] = (_Float16)(re * ci_ + im * cr_);
    }
  }
}

// -- stage3: C[k2][t1][dh] = conj(tw) * (conj(D64) . X)  (inv inner) ----------
__global__ __launch_bounds__(256) void sa_stage3(const _Float16* __restrict__ Xr,
                                                 const _Float16* __restrict__ Xi,
                                                 const _Float16* __restrict__ Dc,
                                                 const _Float16* __restrict__ Ds,
                                                 const _Float16* __restrict__ Dsn,
                                                 const float* __restrict__ twc,
                                                 const float* __restrict__ tws,
                                                 _Float16* __restrict__ Cr,
                                                 _Float16* __restrict__ Ci) {
  __shared__ _Float16 sXr[2][4096], sXi[2][4096];
  __shared__ _Float16 sDc[4096], sDs[4096], sDn[4096];
  int tid = threadIdx.x;
  long batch0 = (long)blockIdx.x * 2;
  if (tid < 32) {
    tdm_load_1row(ldsoff(sXr), Xr + batch0 * 4096, 8192);
    tdm_load_1row(ldsoff(sXi), Xi + batch0 * 4096, 8192);
    __builtin_amdgcn_s_wait_tensorcnt(0);
  }
  for (int c = tid; c < 512; c += 256) async_b128(ldsoff(sDc) + c * 16, Dc + c * 8);
  for (int c = tid; c < 512; c += 256) async_b128(ldsoff(sDs) + c * 16, Ds + c * 8);
  for (int c = tid; c < 512; c += 256) async_b128(ldsoff(sDn) + c * 16, Dsn + c * 8);
  wait_async0();
  __syncthreads();

  int lane = tid & 31, w = tid >> 5;
  int bl = w >> 2;
  int m0 = (w & 3) * 16;          // t1 rows
  long batch = batch0 + bl;
  int k2 = (int)(batch & 63);
  v8f cr[4] = {}, ci[4] = {};
  for (int kb = 0; kb < 64; kb += 32) {          // contract over k1
    v16h aC = fragA_f16(sDc, 64, m0, kb, lane);
    v16h aS = fragA_f16(sDs, 64, m0, kb, lane);
    v16h aN = fragA_f16(sDn, 64, m0, kb, lane);
#pragma unroll
    for (int s = 0; s < 4; ++s) {
      v16h br = fragB_f16strided(sXr[bl], 64, kb, s * 16, lane);
      v16h bi = fragB_f16strided(sXi[bl], 64, kb, s * 16, lane);
      cr[s] = wmma16(aC, br, cr[s]);
      cr[s] = wmma16(aN, bi, cr[s]);
      ci[s] = wmma16(aC, bi, ci[s]);
      ci[s] = wmma16(aS, br, ci[s]);
    }
  }
  int mbase = m0 + ((lane >> 4) << 3);
#pragma unroll
  for (int s = 0; s < 4; ++s) {
    int n = s * 16 + (lane & 15);
#pragma unroll
    for (int r = 0; r < 8; ++r) {
      int t1 = mbase + r;
      float tc = twc[t1 * 64 + k2], ts = tws[t1 * 64 + k2];
      float re = cr[s][r], im = ci[s][r];
      Cr[batch * 4096 + (long)t1 * 64 + n] = (_Float16)(re * tc - im * ts);
      Ci[batch * 4096 + (long)t1 * 64 + n] = (_Float16)(im * tc + re * ts);
    }
  }
}

// -- stage4: y[t][dh] = Re( conj(D64) . C ) = cos.Cr - sin.Ci  (inv outer) ----
__global__ __launch_bounds__(256) void sa_stage4(const _Float16* __restrict__ Cr,
                                                 const _Float16* __restrict__ Ci,
                                                 const _Float16* __restrict__ Dc,
                                                 const _Float16* __restrict__ Dsn,
                                                 _Float16* __restrict__ y_t) {
  __shared__ _Float16 sCr[2][4096], sCi[2][4096];
  __shared__ _Float16 sDc[4096], sDn[4096];
  int tid = threadIdx.x;
  int bh = blockIdx.x >> 5;
  int jp = blockIdx.x & 31;
  const _Float16* gr = Cr + (long)bh * SL_;
  const _Float16* gi = Ci + (long)bh * SL_;
  for (int c = tid; c < 512; c += 256) async_b128(ldsoff(sDc) + c * 16, Dc + c * 8);
  for (int c = tid; c < 512; c += 256) async_b128(ldsoff(sDn) + c * 16, Dsn + c * 8);
  for (int c = tid; c < 1024; c += 256) {
    int jb = c >> 9, r = (c >> 3) & 63, cc = c & 7;
    long gsrc = (long)r * 4096 + jp * 128 + jb * 64 + cc * 8;
    unsigned ldst = (unsigned)((jb * 4096 + r * 64 + cc * 8) * 2);
    async_b128(ldsoff(sCr) + ldst, gr + gsrc);
    async_b128(ldsoff(sCi) + ldst, gi + gsrc);
  }
  wait_async0();
  __syncthreads();

  int lane = tid & 31, w = tid >> 5;
  int jb = w >> 2;
  int m0 = (w & 3) * 16;          // t2 rows
  int j0 = (jp * 2 + jb) * 64;
  v8f acc[4] = {};
  for (int kb = 0; kb < 64; kb += 32) {          // contract over k2
    v16h aC = fragA_f16(sDc, 64, m0, kb, lane);
    v16h aN = fragA_f16(sDn, 64, m0, kb, lane);
#pragma unroll
    for (int s = 0; s < 4; ++s) {
      v16h br = fragB_f16strided(sCr[jb], 64, kb, s * 16, lane);
      v16h bi = fragB_f16strided(sCi[jb], 64, kb, s * 16, lane);
      acc[s] = wmma16(aC, br, acc[s]);
      acc[s] = wmma16(aN, bi, acc[s]);
    }
  }
  int mbase = m0 + ((lane >> 4) << 3);
  _Float16* py = y_t + (long)bh * SL_;
#pragma unroll
  for (int s = 0; s < 4; ++s) {
    int n = j0 + s * 16 + (lane & 15);
#pragma unroll
    for (int r = 0; r < 8; ++r) {
      py[(long)(mbase + r) * 4096 + n] = (_Float16)acc[s][r];
    }
  }
}

// -------------- GEMM2: out = x + y . Wo^T + bo -------------------------------
__global__ __launch_bounds__(256) void sa_gemm2(const _Float16* __restrict__ y_t,
                                                const float* __restrict__ Wo,
                                                const float* __restrict__ bo,
                                                const float* __restrict__ x,
                                                float* __restrict__ out) {
  __shared__ float sW[2][64 * 32];
  int tid = threadIdx.x, lane = tid & 31, w = tid >> 5;
  int wn = blockIdx.x & 15;
  int m0 = ((blockIdx.x >> 4) * 8 + w) * 16;
  int n0 = wn * 64;
  const float* Wb = Wo + (long)n0 * D_;
  for (int c = tid; c < 512; c += 256) {
    int r = c >> 3, cc = c & 7;
    async_b128(ldsoff(sW[0]) + (r * 32 + cc * 4) * 4, Wb + (long)r * D_ + cc * 4);
  }
  wait_async0();
  __syncthreads();
  v8f acc[4] = {};
  int p = 0;
  for (int kb = 0; kb < D_; kb += 32, p ^= 1) {
    if (kb + 32 < D_) {
      for (int c = tid; c < 512; c += 256) {
        int r = c >> 3, cc = c & 7;
        async_b128(ldsoff(sW[p ^ 1]) + (r * 32 + cc * 4) * 4,
                   Wb + (long)r * D_ + kb + 32 + cc * 4);
      }
    }
    v16h a = fragA_ylayout(y_t, m0, kb, lane);
#pragma unroll
    for (int s = 0; s < 4; ++s) {
      v16h b = fragB_f32rows(sW[p], 32, s * 16, 0, lane);
      acc[s] = wmma16(a, b, acc[s]);
    }
    wait_async0();
    __syncthreads();
  }
  int mbase = m0 + ((lane >> 4) << 3);
#pragma unroll
  for (int s = 0; s < 4; ++s) {
    int n = n0 + s * 16 + (lane & 15);
    float bv = bo[n];
#pragma unroll
    for (int r = 0; r < 8; ++r) {
      long m = mbase + r;
      out[m * D_ + n] = x[m * D_ + n] + acc[s][r] + bv;
    }
  }
}

// -----------------------------------------------------------------------------
extern "C" void kernel_launch(void* const* d_in, const int* in_sizes, int n_in,
                              void* d_out, int out_size, void* d_ws, size_t ws_size,
                              hipStream_t stream) {
  const float* x    = (const float*)d_in[0];
  const float* Wqkv = (const float*)d_in[1];
  const float* bqkv = (const float*)d_in[2];
  const float* Wo   = (const float*)d_in[3];
  const float* bo   = (const float*)d_in[4];
  const float* lg   = (const float*)d_in[5];
  const float* ph   = (const float*)d_in[6];
  float* out = (float*)d_out;

  _Float16* v_t = (_Float16*)d_ws;        // v / later reused for y
  _Float16* Ar  = v_t + NE_;
  _Float16* Ai  = Ar + NE_;
  _Float16* Xr  = Ai + NE_;
  _Float16* Xi  = Xr + NE_;
  _Float16* Dc  = Xi + NE_;
  _Float16* Ds  = Dc + 4096;
  _Float16* Dsn = Ds + 4096;
  float* twc = (float*)(Dsn + 4096);
  float* tws = twc + 4096;
  float* fr  = tws + 4096;
  float* fi  = fr + H_ * 4096;

  sa_tables<<<16, 256, 0, stream>>>(Dc, Ds, Dsn, twc, tws);
  sa_filter<<<(H_ * 4096 + 255) / 256, 256, 0, stream>>>(lg, ph, fr, fi);
  sa_gemm1 <<<2048, 256, 0, stream>>>(x, Wqkv, bqkv, v_t);
  sa_stage1<<<2048, 256, 0, stream>>>(v_t, Dc, Dsn, twc, tws, Ar, Ai);
  sa_stage2<<<2048, 256, 0, stream>>>(Ar, Ai, Dc, Ds, Dsn, fr, fi, Xr, Xi);
  sa_stage3<<<2048, 256, 0, stream>>>(Xr, Xi, Dc, Ds, Dsn, twc, tws, Ar, Ai);
  sa_stage4<<<2048, 256, 0, stream>>>(Ar, Ai, Dc, Dsn, v_t);
  sa_gemm2 <<<2048, 256, 0, stream>>>(v_t, Wo, bo, x, out);
}
